// TorchSequenceParallelAttn_58961311039918
// MI455X (gfx1250) — matrix operand order repaired and checked by reading, not compiled, not run on hardware
//
#include <hip/hip_runtime.h>

// Zigzag ring-attention forward shard (rank 1 of 4) for MI455X / gfx1250.
// v4: St = K*Q^T form (P transpose free: C-layout of St == A-layout of P);
// V tiles TRANSPOSED BY THE TDM via a 3-D descriptor walk (tile_dim0=1,
// dim1=keys stride 512, dim2=d stride 1 -> LDS receives Vt[d][key]),
// double-buffered on TENSORcnt; softmax in exp2 domain.
// Per 32-key step: 16 v_wmma_f32_16x16x32_f16, 16 global b128 (K),
// 16 ds_load_b128 (Vt fragments), 1 tensor_load_to_lds.

#define W_SZ   4
#define RANK   1
#define HQ     16
#define HKV    4
#define DH     128
#define SEQ    4096
#define BATCH  2
#define SHARD  1024
#define HALFQ  512
#define KV_TOK_STRIDE (HKV * DH)   // elements between consecutive key tokens
#define TROW   32                  // Vt row = 32 keys (64 B, b128-aligned)

typedef __attribute__((ext_vector_type(16))) _Float16     v16h;
typedef __attribute__((ext_vector_type(8)))  float        v8f;
typedef __attribute__((ext_vector_type(4)))  unsigned int v4u;
typedef __attribute__((ext_vector_type(8)))  int          v8i;
typedef __attribute__((ext_vector_type(4)))  int          v4i;

union H16 {
    v16h v;
    uint4 u[2];
    _Float16 e[16];
};

// TDM: load a 32-key x 128-half V tile (memory row stride 512 halves) into
// LDS *transposed* to Vt[d][key]. 3-D walk: X = 1 element (2 B), Y = 32 keys
// (stride 512 elements), Z = 128 d-positions (stride 1 element). The TDM
// appends each element to LDS in walk order -> LDS layout [d][key], row
// stride 32 halves = 64 B. (Per-element DMA granularity traded for zero
// wave-side transpose work.)
__device__ __forceinline__ void tdm_load_vt(const _Float16* gptr, unsigned lds_off) {
    unsigned long long ga = (unsigned long long)(uintptr_t)gptr;
    v4u g0;
    g0[0] = 1u;                                   // count=1, user descriptor
    g0[1] = lds_off;                              // lds_addr (bytes)
    g0[2] = (unsigned)ga;                         // global_addr[31:0]
    g0[3] = ((unsigned)(ga >> 32) & 0x01FFFFFFu)  // global_addr[56:32]
            | 0x80000000u;                        // type = 2 ("image")
    v8i g1;
    g1[0] = (int)(1u << 16);          // data_size = 1 -> 2 bytes, no pad
    g1[1] = (int)(1u << 16);          // tensor_dim0 = 1   (bits 79:48)
    g1[2] = (int)(32u << 16);         // tensor_dim1 = 32  (bits 111:80)
    g1[3] = (int)(1u << 16);          // tile_dim0   = 1   (bits 127:112)
    g1[4] = (int)(32u | (128u << 16)); // tile_dim1=32 keys | tile_dim2=128 d
    g1[5] = (int)KV_TOK_STRIDE;       // tensor_dim0_stride = 512 (Y step)
    g1[6] = (int)(1u << 16);          // tensor_dim1_stride = 1   (Z step, bits 255:208)
    g1[7] = 0;
    v4i g2;                           // 3-D tensor: group 2 carries dim2
    g2[0] = 128;                      // tensor_dim2 = 128
    g2[1] = 0;                        // tensor_dim3 (iterate off)
    g2[2] = 0;                        // tensor_dim2_stride lo (dim3 unused)
    g2[3] = 0;                        // tile_dim3 = 0 (unused)
    v4i g3; g3[0] = 0; g3[1] = 0; g3[2] = 0; g3[3] = 0;
#if defined(__clang_major__) && (__clang_major__ >= 23)
    v8i gz8; gz8[0]=0; gz8[1]=0; gz8[2]=0; gz8[3]=0; gz8[4]=0; gz8[5]=0; gz8[6]=0; gz8[7]=0;
    __builtin_amdgcn_tensor_load_to_lds(g0, g1, g2, g3, gz8, 0);
#else
    __builtin_amdgcn_tensor_load_to_lds(g0, g1, g2, g3, 0);
#endif
}

// Zigzag block resolution: key position kstep (32-aligned, never straddles a
// 512-token block) -> pointer to that key token's row for (b, hkv).
__device__ __forceinline__ const _Float16*
zig_ptr(const _Float16* base, int b, int hkv, int kstep) {
    const int j  = kstep >> 9;
    const int w  = (j < W_SZ) ? j : (2 * W_SZ - 1 - j);
    const int sh = (j < W_SZ) ? 0 : 1;
    const int rowBase = b * SHARD + sh * HALFQ + (kstep & 511);
    return base + (((size_t)w * (BATCH * SHARD) + rowBase) * HKV + hkv) * DH;
}

__global__ __launch_bounds__(32)
void zigzag_attn_fwd(const _Float16* __restrict__ q,
                     const _Float16* __restrict__ kk,
                     const _Float16* __restrict__ vv,
                     _Float16* __restrict__ out)
{
    // Double-buffered transposed V tile, written only by the TDM.
    __shared__ __align__(16) _Float16 VtLds[2][DH * TROW];

    const int lane = threadIdx.x;       // 0..31 (wave32)
    const int hi   = lane >> 4;
    const int ln   = lane & 15;

    int bid = blockIdx.x;
    const int tile = bid & 31;  bid >>= 5;   // 32 query tiles of 16 per half
    const int half = bid & 1;   bid >>= 1;
    const int h    = bid & 15;  bid >>= 4;
    const int b    = bid;

    const int hkv   = h >> 2;                        // G = HQ/HKV = 4
    const int qbase = tile * 16;
    const int off   = half ? (SEQ - HALFQ * (RANK + 1)) : (HALFQ * RANK);
    const int kend  = qbase + 16 + off;              // exclusive causal bound
    const int nsteps = (kend + 31) >> 5;

    // 1/sqrt(D) * log2(e): folded into Q so logits are exp2-domain-ready.
    const _Float16 cs = (_Float16)(0.08838834764831845f * 1.4426950408889634f);

    // ---- Preload Q as B-fragments for St = K * Q^T (pre-scaled).
    // B layout (16-bit, 32x16): lane col n=ln (query), elem e<8 -> k(d) = 8*hi+e, e>=8 -> 16+8*hi+(e-8).
    const int qrow = b * SHARD + half * HALFQ + qbase + ln;
    const _Float16* qp = q + ((size_t)qrow * HQ + h) * DH;
    H16 qb[4];
#pragma unroll
    for (int c = 0; c < 4; ++c) {
        qb[c].u[0] = *(const uint4*)(qp + c * 32 + 8 * hi);
        qb[c].u[1] = *(const uint4*)(qp + c * 32 + 16 + 8 * hi);
        qb[c].v = qb[c].v * cs;
    }

    // Per-lane flash state (exp2 domain) for query column q = ln.
    float mstat = -__builtin_inff();
    float lstat = 0.0f;
    v8f oacc[8];                     // O C-layout: row m(query)=8*hi+r, col=d
#pragma unroll
    for (int dt = 0; dt < 8; ++dt) { v8f z = {}; oacc[dt] = z; }

    const unsigned ldsoff[2] = { (unsigned)(uintptr_t)&VtLds[0][0],
                                 (unsigned)(uintptr_t)&VtLds[1][0] };

    // Prime the pipeline: TDM transpose-fetch of V tile for step 0.
    tdm_load_vt(zig_ptr(vv, b, hkv, 0), ldsoff[0]);

    for (int s = 0; s < nsteps; ++s) {
        const int kstep = s * 32;
        const _Float16* kbp = zig_ptr(kk, b, hkv, kstep);

        // ---- St = K * Q^T: A = K rows (A layout: lane m=ln -> key row,
        // elems: e<8 -> d=c*32+8*hi+e, e>=8 -> d=c*32+16+8*hi+(e-8)).
        const _Float16* krow0 = kbp + (size_t)ln * KV_TOK_STRIDE;
        const _Float16* krow1 = kbp + (size_t)(16 + ln) * KV_TOK_STRIDE;
        v8f st0 = {}, st1 = {};
#pragma unroll
        for (int c = 0; c < 4; ++c) {
            H16 a0, a1;
            a0.u[0] = *(const uint4*)(krow0 + c * 32 + 8 * hi);
            a0.u[1] = *(const uint4*)(krow0 + c * 32 + 16 + 8 * hi);
            a1.u[0] = *(const uint4*)(krow1 + c * 32 + 8 * hi);
            a1.u[1] = *(const uint4*)(krow1 + c * 32 + 16 + 8 * hi);
            st0 = __builtin_amdgcn_wmma_f32_16x16x32_f16(false, a0.v, false, qb[c].v,
                                                         (short)0, st0, false, false);
            st1 = __builtin_amdgcn_wmma_f32_16x16x32_f16(false, a1.v, false, qb[c].v,
                                                         (short)0, st1, false, false);
        }

        // ---- Online softmax, exp2 domain, per query column.
        // Lane holds St[key][q=ln]: keys kstep+8*hi+e (st0) and +16 (st1).
        const int kb  = kstep + 8 * hi;
        const int lim = qbase + ln + off;            // causal bound for query ln
        float x[16];
#pragma unroll
        for (int e = 0; e < 8; ++e) {
            x[e]     = (kb + e      <= lim) ? st0[e] : -__builtin_inff();
            x[e + 8] = (kb + 16 + e <= lim) ? st1[e] : -__builtin_inff();
        }
        float mx = x[0];
#pragma unroll
        for (int e = 1; e < 16; ++e) mx = fmaxf(mx, x[e]);
        mx = fmaxf(mx, __shfl_xor(mx, 16, 32));
        const float mnew = fmaxf(mstat, mx);
        const float a = __builtin_amdgcn_exp2f(mstat - mnew);   // alpha
        float rs = 0.0f;
        H16 pa;                                      // P A-fragment, in-lane!
#pragma unroll
        for (int e = 0; e < 16; ++e) {
            const float p = __builtin_amdgcn_exp2f(x[e] - mnew);
            rs += p;
            pa.e[e] = (_Float16)p;
        }
        rs += __shfl_xor(rs, 16, 32);
        lstat = lstat * a + rs;
        mstat = mnew;

        // Broadcast alpha to the O-accumulator rows (query m = 8*hi+r lives
        // in lane m's softmax state) and rescale.
        float am[8];
#pragma unroll
        for (int r = 0; r < 8; ++r) am[r] = __shfl(a, 8 * hi + r, 32);
#pragma unroll
        for (int dt = 0; dt < 8; ++dt)
#pragma unroll
            for (int r = 0; r < 8; ++r)
                oacc[dt][r] *= am[r];

        // ---- Overlap: issue TDM for the next V tile, then wait for this one.
        if (s + 1 < nsteps) {
            tdm_load_vt(zig_ptr(vv, b, hkv, kstep + 32), ldsoff[(s + 1) & 1]);
            __builtin_amdgcn_s_wait_tensorcnt((short)1);
        } else {
            __builtin_amdgcn_s_wait_tensorcnt((short)0);
        }
        asm volatile("" ::: "memory");

        // ---- O += P * V. B fragment of V from transposed LDS: lane col
        // d = dt*16+ln, elem e<8 -> key = 8*hi+e, e>=8 -> key = 16+8*hi+(e-8)
        // Vt[d][8*hi ..] and Vt[d][16+8*hi ..] -> two aligned ds_load_b128 per tile.
        const _Float16* vbase = &VtLds[s & 1][8 * hi];
#pragma unroll
        for (int dt = 0; dt < 8; ++dt) {
            H16 bv;
            const _Float16* vp = vbase + (dt * 16 + ln) * TROW;
            bv.u[0] = *(const uint4*)(vp);
            bv.u[1] = *(const uint4*)(vp + 16);
            oacc[dt] = __builtin_amdgcn_wmma_f32_16x16x32_f16(false, pa.v, false, bv.v,
                                                              (short)0, oacc[dt], false, false);
        }
    }

    // ---- Epilogue: fetch per-query denominators, normalize, store f16.
    float lm[8];
#pragma unroll
    for (int r = 0; r < 8; ++r) lm[r] = __shfl(lstat, 8 * hi + r, 32);
    const int orow0 = b * SHARD + half * HALFQ + qbase;
#pragma unroll
    for (int r = 0; r < 8; ++r) {
        const float inv = 1.0f / lm[r];
        _Float16* op = out + ((size_t)(orow0 + 8 * hi + r) * HQ + h) * DH + ln;
#pragma unroll
        for (int dt = 0; dt < 8; ++dt)
            op[dt * 16] = (_Float16)(oacc[dt][r] * inv);
    }
}

extern "C" void kernel_launch(void* const* d_in, const int* in_sizes, int n_in,
                              void* d_out, int out_size, void* d_ws, size_t ws_size,
                              hipStream_t stream) {
    const _Float16* q  = (const _Float16*)d_in[0];   // (B*shard, HQ, D) f16
    const _Float16* kk = (const _Float16*)d_in[1];   // (W, B*shard, HKV, D) f16
    const _Float16* vr = (const _Float16*)d_in[2];   // (W, B*shard, HKV, D) f16
    _Float16* out = (_Float16*)d_out;                // (B*shard, HQ, D) f16

    const int nblocks = BATCH * HQ * 2 * (HALFQ / 16);   // 2048
    zigzag_attn_fwd<<<dim3(nblocks), dim3(32), 0, stream>>>(q, kk, vr, out);
}
